// TemporalRNN_14628658610812
// MI455X (gfx1250) — matrix operand order repaired and checked by reading, not compiled, not run on hardware
//
#include <hip/hip_runtime.h>
#include <hip/hip_bf16.h>

#define B_   32768
#define T_   30
#define I_   13
#define H_   128
#define G_   512      // 4*H
#define BT   64       // batch rows per block
#define NW   8        // waves per block (256 threads, wave32)

typedef __attribute__((ext_vector_type(16))) __bf16 v16bf;
typedef __attribute__((ext_vector_type(8)))  __bf16 v8bf;
typedef __attribute__((ext_vector_type(8)))  float  v8f;

__device__ __forceinline__ __bf16 f2bf(float f) {
    union { float f; unsigned u; } v; v.f = f;
    unsigned r = (v.u + 0x7FFFu + ((v.u >> 16) & 1u)) >> 16;   // RNE
    unsigned short h = (unsigned short)r;
    __bf16 b; __builtin_memcpy(&b, &h, 2);
    return b;
}

__device__ __forceinline__ v16bf cat16(v8bf lo, v8bf hi) {
    return __builtin_shufflevector(lo, hi, 0,1,2,3,4,5,6,7,8,9,10,11,12,13,14,15);
}

__device__ __forceinline__ v8f wmma_bf16(v16bf a, v16bf b, v8f c) {
    return __builtin_amdgcn_wmma_f32_16x16x32_bf16(false, a, false, b, (short)0, c, false, false);
}

// A-fragment (16x32 bf16, M x K) from LDS, row stride = H_ (128).
// ISA layout: lanes 0-15 row m: K=k0+0..7 (V0-3), K=k0+16..23 (V4-7);
//             lanes 16-31 row m: K=k0+8..15,       K=k0+24..31.
__device__ __forceinline__ v16bf afrag_lds(const __bf16* Hs, int lrow, int k0, int lane) {
    int off = (lane & 16) ? 8 : 0;
    const __bf16* p = Hs + lrow * H_ + k0 + off;
    v8bf lo = *(const v8bf*)p;
    v8bf hi = *(const v8bf*)(p + 16);
    return cat16(lo, hi);
}

// B-fragment (32x16 bf16, K x N) from global, W stored [N][K] row-major, stride=128.
// Lane n (0-15): column n, K=k0+0..7,16..23; lane n+16: column n, K=k0+8..15,24..31.
__device__ __forceinline__ v16bf bfrag_g(const __bf16* W, int col, int k0, int lane) {
    int kk = k0 + ((lane & 16) ? 8 : 0);
    const __bf16* p = W + (size_t)col * 128 + kk;
    v8bf lo = *(const v8bf*)p;
    v8bf hi = *(const v8bf*)(p + 16);
    return cat16(lo, hi);
}

// Same but stride 32 (padded Wih0: [512][32])
__device__ __forceinline__ v16bf bfrag_g32(const __bf16* W, int col, int lane) {
    int kk = (lane & 16) ? 8 : 0;
    const __bf16* p = W + (size_t)col * 32 + kk;
    v8bf lo = *(const v8bf*)p;
    v8bf hi = *(const v8bf*)(p + 16);
    return cat16(lo, hi);
}

// C/D f32 16x16 layout: VGPR r, lane l -> row = r + 8*(l>=16), col = l%16
__device__ __forceinline__ void store_acc_lds(float* gb, int ldg, int m0, int n0, int lane, v8f acc) {
    int col  = n0 * 16 + (lane & 15);
    int rb   = m0 * 16 + ((lane & 16) ? 8 : 0);
    #pragma unroll
    for (int r = 0; r < 8; ++r) gb[(rb + r) * ldg + col] = acc[r];
}

// branchless activations: native v_exp_f32 + v_rcp_f32, correct saturation at +-inf
__device__ __forceinline__ float sigm(float x) {
    return __builtin_amdgcn_rcpf(1.f + __expf(-x));
}
__device__ __forceinline__ float tanhx(float x) {
    return 1.f - 2.f * __builtin_amdgcn_rcpf(__expf(2.f * x) + 1.f);
}

// ---------------------------------------------------------------- pack kernels
__global__ void k_pack_x(const float* __restrict__ x, __bf16* __restrict__ xp) {
    int i = blockIdx.x * blockDim.x + threadIdx.x;     // over B*T
    if (i >= B_ * T_) return;
    const float* s = x + (size_t)i * I_;
    __bf16* d = xp + (size_t)i * 16;
    #pragma unroll
    for (int k = 0; k < 16; ++k) d[k] = f2bf(k < I_ ? s[k] : 0.f);
}

__global__ void k_cvt_bf16(const float* __restrict__ s, __bf16* __restrict__ d, int n) {
    int i = blockIdx.x * blockDim.x + threadIdx.x;
    if (i < n) d[i] = f2bf(s[i]);
}

__global__ void k_pack_wih0(const float* __restrict__ s, __bf16* __restrict__ d) {
    int i = blockIdx.x * blockDim.x + threadIdx.x;     // over 512*32
    if (i >= G_ * 32) return;
    int r = i >> 5, k = i & 31;
    d[i] = f2bf(k < I_ ? s[r * I_ + k] : 0.f);
}

__global__ void k_bias_sum(const float* __restrict__ bi0, const float* __restrict__ bh0,
                           const float* __restrict__ bi1, const float* __restrict__ bh1,
                           float* __restrict__ bs0, float* __restrict__ bs1) {
    int i = blockIdx.x * blockDim.x + threadIdx.x;
    if (i < G_) { bs0[i] = bi0[i] + bh0[i]; bs1[i] = bi1[i] + bh1[i]; }
}

// ---------------------------------------------------------------- fused LSTM
__global__ void __launch_bounds__(256)
k_lstm_fused(const __bf16* __restrict__ xp,     // [B][T][16]
             const __bf16* __restrict__ Wih0p,  // [512][32]
             const __bf16* __restrict__ Whh0p,  // [512][128]
             const __bf16* __restrict__ Wih1p,  // [512][128]
             const __bf16* __restrict__ Whh1p,  // [512][128]
             const __bf16* __restrict__ fcWp,   // [128][128]
             const float* __restrict__ bs0,     // bih0+bhh0  [512]
             const float* __restrict__ bs1,     // bih1+bhh1  [512]
             float* __restrict__ ybuf)          // [B][128]
{
    extern __shared__ char smem[];
    float*  gbuf = (float*)smem;                       // 64*512 f32 = 128 KB
    __bf16* h1s  = (__bf16*)(smem + BT * G_ * 4);      // 64*128 bf16 = 16 KB
    __bf16* h2s  = h1s + BT * H_;                      // 16 KB

    const int tid     = threadIdx.x;
    const int lane    = tid & 31;
    const int wave    = __builtin_amdgcn_readfirstlane(tid >> 5);
    const int rowbase = blockIdx.x * BT;

    for (int i = tid; i < BT * H_; i += 256) { h1s[i] = f2bf(0.f); h2s[i] = f2bf(0.f); }

    // per-thread cell state: row = tid>>2 (0..63), cols = (tid&3)*32 + j
    float c1v[32], c2v[32];
    #pragma unroll
    for (int j = 0; j < 32; ++j) { c1v[j] = 0.f; c2v[j] = 0.f; }
    const int urow  = tid >> 2;
    const int ucol0 = (tid & 3) * 32;
    __syncthreads();

    for (int t = 0; t < T_; ++t) {
        // ---- phase 1: layer-0 gates  gbuf = x(t)@Wih0^T + h1(t-1)@Whh0^T + bias0
        #pragma unroll 1
        for (int it = 0; it < 16; ++it) {               // 128 tiles / 8 waves
            int tile = it * NW + wave;
            int m0 = tile & 3, n0 = tile >> 2;
            int arow = m0 * 16 + (lane & 15);
            int bcol = n0 * 16 + (lane & 15);

            // hoisted fragments (one load clause, then WMMA chain)
            const __bf16* px = xp + ((size_t)(rowbase + arow) * T_ + t) * 16
                                  + ((lane & 16) ? 8 : 0);
            v8bf  xlo = *(const v8bf*)px;
            v8bf  zz  = {};
            v16bf ax  = cat16(xlo, zz);
            v16bf bx  = bfrag_g32(Wih0p, bcol, lane);
            v16bf a0  = afrag_lds(h1s, arow,  0, lane);
            v16bf a1  = afrag_lds(h1s, arow, 32, lane);
            v16bf a2  = afrag_lds(h1s, arow, 64, lane);
            v16bf a3  = afrag_lds(h1s, arow, 96, lane);
            v16bf b0  = bfrag_g(Whh0p, bcol,  0, lane);
            v16bf b1  = bfrag_g(Whh0p, bcol, 32, lane);
            v16bf b2  = bfrag_g(Whh0p, bcol, 64, lane);
            v16bf b3  = bfrag_g(Whh0p, bcol, 96, lane);

            float bv = bs0[bcol];                       // bias: same col for all 8 rows
            v8f acc = {bv, bv, bv, bv, bv, bv, bv, bv};
            acc = wmma_bf16(ax, bx, acc);
            acc = wmma_bf16(a0, b0, acc);
            acc = wmma_bf16(a1, b1, acc);
            acc = wmma_bf16(a2, b2, acc);
            acc = wmma_bf16(a3, b3, acc);
            store_acc_lds(gbuf, G_, m0, n0, lane, acc);
        }
        __syncthreads();

        // ---- phase 2: layer-0 cell update, write h1(t)  (fully unrolled: static
        //      register addressing of c1v[], no v_movrels)
        #pragma unroll
        for (int j = 0; j < 32; ++j) {
            int col = ucol0 + j;
            float ig = gbuf[urow * G_ +       col];
            float fg = gbuf[urow * G_ + 128 + col];
            float gg = gbuf[urow * G_ + 256 + col];
            float og = gbuf[urow * G_ + 384 + col];
            c1v[j] = sigm(fg) * c1v[j] + sigm(ig) * tanhx(gg);
            h1s[urow * H_ + col] = f2bf(sigm(og) * tanhx(c1v[j]));
        }
        __syncthreads();

        // ---- phase 3: layer-1 gates  gbuf = h1(t)@Wih1^T + h2(t-1)@Whh1^T + bias1
        #pragma unroll 1
        for (int it = 0; it < 16; ++it) {
            int tile = it * NW + wave;
            int m0 = tile & 3, n0 = tile >> 2;
            int arow = m0 * 16 + (lane & 15);
            int bcol = n0 * 16 + (lane & 15);

            float bv = bs1[bcol];
            v8f acc = {bv, bv, bv, bv, bv, bv, bv, bv};
            {
                v16bf a0 = afrag_lds(h1s, arow,  0, lane);
                v16bf a1 = afrag_lds(h1s, arow, 32, lane);
                v16bf a2 = afrag_lds(h1s, arow, 64, lane);
                v16bf a3 = afrag_lds(h1s, arow, 96, lane);
                v16bf b0 = bfrag_g(Wih1p, bcol,  0, lane);
                v16bf b1 = bfrag_g(Wih1p, bcol, 32, lane);
                v16bf b2 = bfrag_g(Wih1p, bcol, 64, lane);
                v16bf b3 = bfrag_g(Wih1p, bcol, 96, lane);
                acc = wmma_bf16(a0, b0, acc);
                acc = wmma_bf16(a1, b1, acc);
                acc = wmma_bf16(a2, b2, acc);
                acc = wmma_bf16(a3, b3, acc);
            }
            {
                v16bf a0 = afrag_lds(h2s, arow,  0, lane);
                v16bf a1 = afrag_lds(h2s, arow, 32, lane);
                v16bf a2 = afrag_lds(h2s, arow, 64, lane);
                v16bf a3 = afrag_lds(h2s, arow, 96, lane);
                v16bf b0 = bfrag_g(Whh1p, bcol,  0, lane);
                v16bf b1 = bfrag_g(Whh1p, bcol, 32, lane);
                v16bf b2 = bfrag_g(Whh1p, bcol, 64, lane);
                v16bf b3 = bfrag_g(Whh1p, bcol, 96, lane);
                acc = wmma_bf16(a0, b0, acc);
                acc = wmma_bf16(a1, b1, acc);
                acc = wmma_bf16(a2, b2, acc);
                acc = wmma_bf16(a3, b3, acc);
            }
            store_acc_lds(gbuf, G_, m0, n0, lane, acc);
        }
        __syncthreads();

        // ---- phase 4: layer-1 cell update, write h2(t)
        #pragma unroll
        for (int j = 0; j < 32; ++j) {
            int col = ucol0 + j;
            float ig = gbuf[urow * G_ +       col];
            float fg = gbuf[urow * G_ + 128 + col];
            float gg = gbuf[urow * G_ + 256 + col];
            float og = gbuf[urow * G_ + 384 + col];
            c2v[j] = sigm(fg) * c2v[j] + sigm(ig) * tanhx(gg);
            h2s[urow * H_ + col] = f2bf(sigm(og) * tanhx(c2v[j]));
        }
        __syncthreads();
    }

    // ---- FC: y = h2(T-1) @ fcW^T   (fcb cancels under batch-norm)
    #pragma unroll 1
    for (int it = 0; it < 4; ++it) {                    // 32 tiles / 8 waves
        int tile = it * NW + wave;
        int m0 = tile & 3, n0 = tile >> 2;
        int arow = m0 * 16 + (lane & 15);
        int bcol = n0 * 16 + (lane & 15);
        v16bf a0 = afrag_lds(h2s, arow,  0, lane);
        v16bf a1 = afrag_lds(h2s, arow, 32, lane);
        v16bf a2 = afrag_lds(h2s, arow, 64, lane);
        v16bf a3 = afrag_lds(h2s, arow, 96, lane);
        v16bf b0 = bfrag_g(fcWp, bcol,  0, lane);
        v16bf b1 = bfrag_g(fcWp, bcol, 32, lane);
        v16bf b2 = bfrag_g(fcWp, bcol, 64, lane);
        v16bf b3 = bfrag_g(fcWp, bcol, 96, lane);
        v8f acc = {};
        acc = wmma_bf16(a0, b0, acc);
        acc = wmma_bf16(a1, b1, acc);
        acc = wmma_bf16(a2, b2, acc);
        acc = wmma_bf16(a3, b3, acc);
        int rb = rowbase + m0 * 16 + ((lane & 16) ? 8 : 0);
        #pragma unroll
        for (int r = 0; r < 8; ++r) ybuf[(size_t)(rb + r) * H_ + bcol] = acc[r];
    }
}

// ---------------------------------------------------------------- batch-norm
__global__ void k_bn_stats(const float* __restrict__ y, float* __restrict__ stats) {
    int col = blockIdx.x;            // 0..127
    __shared__ float ssum[256], ssq[256];
    float s = 0.f, q = 0.f;
    for (int r = threadIdx.x; r < B_; r += 256) {
        float v = y[(size_t)r * H_ + col];
        s += v; q += v * v;
    }
    ssum[threadIdx.x] = s; ssq[threadIdx.x] = q;
    __syncthreads();
    for (int off = 128; off > 0; off >>= 1) {
        if (threadIdx.x < off) {
            ssum[threadIdx.x] += ssum[threadIdx.x + off];
            ssq [threadIdx.x] += ssq [threadIdx.x + off];
        }
        __syncthreads();
    }
    if (threadIdx.x == 0) {
        float mean = ssum[0] / (float)B_;
        float var  = ssq[0] / (float)B_ - mean * mean;
        stats[col]       = mean;
        stats[128 + col] = rsqrtf(var + 1e-5f);
    }
}

__global__ void k_bn_apply(const float* __restrict__ y, const float* __restrict__ stats,
                           const float* __restrict__ gamma, const float* __restrict__ beta,
                           float* __restrict__ out) {
    int i = blockIdx.x * blockDim.x + threadIdx.x;
    if (i >= B_ * H_) return;
    int col = i & (H_ - 1);
    out[i] = gamma[col] * (y[i] - stats[col]) * stats[128 + col] + beta[col];
}

// ---------------------------------------------------------------- launcher
extern "C" void kernel_launch(void* const* d_in, const int* in_sizes, int n_in,
                              void* d_out, int out_size, void* d_ws, size_t ws_size,
                              hipStream_t stream) {
    (void)in_sizes; (void)n_in; (void)out_size; (void)ws_size;
    const float* x    = (const float*)d_in[0];
    const float* Wih0 = (const float*)d_in[1];
    const float* Whh0 = (const float*)d_in[2];
    const float* bih0 = (const float*)d_in[3];
    const float* bhh0 = (const float*)d_in[4];
    const float* Wih1 = (const float*)d_in[5];
    const float* Whh1 = (const float*)d_in[6];
    const float* bih1 = (const float*)d_in[7];
    const float* bhh1 = (const float*)d_in[8];
    const float* fcW  = (const float*)d_in[9];
    const float* gamma= (const float*)d_in[11];
    const float* beta = (const float*)d_in[12];
    float* out = (float*)d_out;

    size_t off = 0;
    auto carve = [&](size_t bytes) {
        void* p = (char*)d_ws + off;
        off += (bytes + 255) & ~(size_t)255;
        return p;
    };
    __bf16* xp    = (__bf16*)carve((size_t)B_ * T_ * 16 * 2);
    __bf16* Wih0p = (__bf16*)carve((size_t)G_ * 32 * 2);
    __bf16* Whh0p = (__bf16*)carve((size_t)G_ * H_ * 2);
    __bf16* Wih1p = (__bf16*)carve((size_t)G_ * H_ * 2);
    __bf16* Whh1p = (__bf16*)carve((size_t)G_ * H_ * 2);
    __bf16* fcWp  = (__bf16*)carve((size_t)H_ * H_ * 2);
    float*  bs0   = (float*) carve((size_t)G_ * 4);
    float*  bs1   = (float*) carve((size_t)G_ * 4);
    float*  ybuf  = (float*) carve((size_t)B_ * H_ * 4);
    float*  stats = (float*) carve(256 * 4);

    k_pack_x   <<<(B_ * T_ + 255) / 256, 256, 0, stream>>>(x, xp);
    k_pack_wih0<<<(G_ * 32 + 255) / 256, 256, 0, stream>>>(Wih0, Wih0p);
    k_cvt_bf16 <<<(G_ * H_ + 255) / 256, 256, 0, stream>>>(Whh0, Whh0p, G_ * H_);
    k_cvt_bf16 <<<(G_ * H_ + 255) / 256, 256, 0, stream>>>(Wih1, Wih1p, G_ * H_);
    k_cvt_bf16 <<<(G_ * H_ + 255) / 256, 256, 0, stream>>>(Whh1, Whh1p, G_ * H_);
    k_cvt_bf16 <<<(H_ * H_ + 255) / 256, 256, 0, stream>>>(fcW,  fcWp,  H_ * H_);
    k_bias_sum <<<(G_ + 255) / 256, 256, 0, stream>>>(bih0, bhh0, bih1, bhh1, bs0, bs1);

    size_t lds = (size_t)BT * G_ * 4 + 2 * (size_t)BT * H_ * 2;   // 160 KB
    k_lstm_fused<<<B_ / BT, 256, lds, stream>>>(xp, Wih0p, Whh0p, Wih1p, Whh1p, fcWp,
                                                bs0, bs1, ybuf);

    k_bn_stats<<<H_, 256, 0, stream>>>(ybuf, stats);
    k_bn_apply<<<(B_ * H_ + 255) / 256, 256, 0, stream>>>(ybuf, stats, gamma, beta, out);
}
